// GatedGraphConvolution_67439576481818
// MI455X (gfx1250) — matrix operand order repaired and checked by reading, not compiled, not run on hardware
//
#include <hip/hip_runtime.h>
#include <hip/hip_bf16.h>

// ---------------------------------------------------------------------------
// GatedGraphConvolution for MI455X (gfx1250, wave32, WMMA bf16 16x16x32).
//
//  support = x @ w1                      (bf16 WMMA, f32 accum)
//  acc     = eps*support + b1            (seed for SpMM)
//  trans   = sigmoid(res @ w2 + b2)      (stored in d_out[0:NN) as scratch)
//  gate1   = x @ w3 + b3                 (stored in d_out[NN:2NN) as scratch)
//  acc    += A_coo @ support             (row-segmented SpMM, f32 atomics)
//  output  = relu(acc)
//  gate    = sigmoid(gate1 + output@w4 + b4)
//  out1    = output + gate*(trans-output)   -> overwrites trans slot
//  out2    = trans  + gate*(output-trans)   -> overwrites gate1 slot
// ---------------------------------------------------------------------------

typedef __attribute__((ext_vector_type(16))) __bf16 bf16x16;
typedef __attribute__((ext_vector_type(8)))  __bf16 bf16x8;
typedef __attribute__((ext_vector_type(8)))  float  v8f;

#define DIM 128
#define LDSW_STRIDE 136   // bf16 elems; 272B column pitch -> conflict-free ds_load_b128
#define WAVES_PER_BLOCK 8
#define EPW 128           // edges per wave in SpMM

__device__ __forceinline__ float sigmoidf_fast(float x) {
    return 1.0f / (1.0f + __expf(-x));
}

// Stage W (f32 row-major [128x128]) into LDS as bf16 column-major with pad.
__device__ __forceinline__ void stage_weights(const float* __restrict__ W,
                                              __bf16* ldsW, int tid) {
    for (int idx = tid; idx < DIM * DIM; idx += 256) {
        const int k = idx >> 7;       // row of W (coalesced global read)
        const int c = idx & 127;      // col of W
        ldsW[c * LDSW_STRIDE + k] = (__bf16)W[idx];
    }
    __syncthreads();
}

// B fragment (32x16 K-tile of W), same K order as A:
// lane holds K chunks [kg*8,+8) and [kg*8+16,+8) of the kb*32 tile.
__device__ __forceinline__ bf16x16 load_bfrag(const __bf16* ldsW, int nb,
                                              int kb, int nlane, int kg) {
    const __bf16* bp = ldsW + (nb * 16 + nlane) * LDSW_STRIDE + kb * 32 + kg * 8;
    bf16x8 lo = *(const bf16x8*)bp;        // 16B aligned -> ds_load_b128
    bf16x8 hi = *(const bf16x8*)(bp + 16);
    bf16x16 b;
#pragma unroll
    for (int i = 0; i < 8; ++i) { b[i] = lo[i]; b[8 + i] = hi[i]; }
    return b;
}

// Preload all 16 float4s of one 16-row A tile (one load clause), then convert
// pairwise (adjacent regs -> v_cvt_pk_bf16_f32).
// A layout (16x32 bf16): lane holds row m0+nlane, chunks [kg*8,+8) and [kg*8+16,+8).
__device__ __forceinline__ void load_afrags(const float* __restrict__ Xrow,
                                            bf16x16 a[4], bool relu_in) {
    float4 raw[16];
#pragma unroll
    for (int kb = 0; kb < 4; ++kb) {
        const float* p = Xrow + kb * 32;
        raw[kb * 4 + 0] = *(const float4*)(p);        // chunk0 [0..3]
        raw[kb * 4 + 1] = *(const float4*)(p + 4);    // chunk0 [4..7]
        raw[kb * 4 + 2] = *(const float4*)(p + 16);   // chunk1 [0..3]
        raw[kb * 4 + 3] = *(const float4*)(p + 20);   // chunk1 [4..7]
    }
#pragma unroll
    for (int kb = 0; kb < 4; ++kb) {
#pragma unroll
        for (int q = 0; q < 4; ++q) {
            float4 f = raw[kb * 4 + q];
            if (relu_in) {
                f.x = fmaxf(f.x, 0.f); f.y = fmaxf(f.y, 0.f);
                f.z = fmaxf(f.z, 0.f); f.w = fmaxf(f.w, 0.f);
            }
            const int base = (q >> 1) * 8 + (q & 1) * 4;  // 0,4 -> chunk0; 8,12 -> chunk1
            a[kb][base + 0] = (__bf16)f.x;
            a[kb][base + 1] = (__bf16)f.y;
            a[kb][base + 2] = (__bf16)f.z;
            a[kb][base + 3] = (__bf16)f.w;
        }
    }
}

// ---------------------------------------------------------------------------
// Generic node GEMM: 16-row tile per wave, all 128 output cols, K=128.
// mode 0: out = X@W (support); out2 = eps*(X@W) + bias   (acc seed, bias=b1)
// mode 1: out = sigmoid(X@W + bias)                      (trans)
// mode 2: out = X@W + bias                               (gate1)
// ---------------------------------------------------------------------------
__global__ __launch_bounds__(256)
__attribute__((amdgpu_waves_per_eu(2)))
void gemm_node(
    const float* __restrict__ X, const float* __restrict__ W,
    const float* __restrict__ bias, const float* __restrict__ eps_p,
    float* __restrict__ out, float* __restrict__ out2,
    int ntiles, int mode)
{
    __shared__ __bf16 ldsW[DIM * LDSW_STRIDE];
    const int tid = threadIdx.x;
    stage_weights(W, ldsW, tid);

    const int wave = tid >> 5;
    const int lane = tid & 31;
    const int tile = blockIdx.x * WAVES_PER_BLOCK + wave;
    if (tile >= ntiles) return;

    const int m0    = tile * 16;
    const int nlane = lane & 15;   // M index for A-frag, N index for B/C-frag
    const int kg    = lane >> 4;   // half-wave K group

    bf16x16 a[4];
    load_afrags(X + (size_t)(m0 + nlane) * DIM + kg * 8, a, false);

    v8f acc[8] = {};
#pragma unroll
    for (int kb = 0; kb < 4; ++kb) {
#pragma unroll
        for (int nb = 0; nb < 8; ++nb) {
            bf16x16 b = load_bfrag(ldsW, nb, kb, nlane, kg);
            acc[nb] = __builtin_amdgcn_wmma_f32_16x16x32_bf16(
                false, a[kb], false, b, (short)0, acc[nb], false, false);
        }
    }

    const float eps = (mode == 0) ? eps_p[0] : 0.0f;
#pragma unroll
    for (int nb = 0; nb < 8; ++nb) {
        const int col = nb * 16 + nlane;
        const float bv = bias[col];
#pragma unroll
        for (int r = 0; r < 8; ++r) {
            const int m = r + (kg << 3);                 // C/D: lanes16-31 -> M+8
            const size_t idx = (size_t)(m0 + m) * DIM + col;
            const float s = acc[nb][r];
            if (mode == 0)      { out[idx] = s; out2[idx] = eps * s + bv; }
            else if (mode == 1) { out[idx] = sigmoidf_fast(s + bv); }
            else                { out[idx] = s + bv; }
        }
    }
}

// ---------------------------------------------------------------------------
// Row-segmented COO SpMM: acc[row] += vals * support[col]; rows sorted.
// One wave per EPW edges; lane owns 4 features; atomics only on row change.
// support (51MB) stays resident in the 192MB L2 -> gathers hit L2.
// ---------------------------------------------------------------------------
__global__ __launch_bounds__(256) void spmm_coo(
    const int* __restrict__ adj_row, const int* __restrict__ adj_col,
    const float* __restrict__ adj_vals, const float* __restrict__ support,
    float* __restrict__ accbuf, int nedges)
{
    const int lane = threadIdx.x & 31;
    const long wave = (long)blockIdx.x * WAVES_PER_BLOCK + (threadIdx.x >> 5);
    const long e0 = wave * EPW;
    if (e0 >= nedges) return;
    const long e1 = min(e0 + (long)EPW, (long)nedges);

    int cur_row = adj_row[e0];
    float ax = 0.f, ay = 0.f, az = 0.f, aw = 0.f;

    for (long e = e0; e < e1; ++e) {
        const int r = adj_row[e];
        if (r != cur_row) {
            float* dst = accbuf + (size_t)cur_row * DIM + lane * 4;
            atomicAdd(dst + 0, ax); atomicAdd(dst + 1, ay);
            atomicAdd(dst + 2, az); atomicAdd(dst + 3, aw);
            ax = ay = az = aw = 0.f;
            cur_row = r;
        }
        const int   c = adj_col[e];
        const float v = adj_vals[e];
        if (e + 1 < e1)  // one edge ahead -> global_prefetch_b8
            __builtin_prefetch(support + (size_t)adj_col[e + 1] * DIM + lane * 4, 0, 1);
        const float4 s = *((const float4*)(support + (size_t)c * DIM) + lane);
        ax = fmaf(v, s.x, ax); ay = fmaf(v, s.y, ay);
        az = fmaf(v, s.z, az); aw = fmaf(v, s.w, aw);
    }
    float* dst = accbuf + (size_t)cur_row * DIM + lane * 4;
    atomicAdd(dst + 0, ax); atomicAdd(dst + 1, ay);
    atomicAdd(dst + 2, az); atomicAdd(dst + 3, aw);
}

// ---------------------------------------------------------------------------
// Finalize: output=relu(acc); gate2=output@w4+b4 (WMMA); gated blend.
// trans_io aliases d_out[0:NN) (holds trans, receives out1);
// gate_io  aliases d_out[NN:2NN) (holds gate1, receives out2).
// Each element is read then written by exactly one thread -> no hazard.
// ---------------------------------------------------------------------------
__global__ __launch_bounds__(256)
__attribute__((amdgpu_waves_per_eu(2)))
void finalize_gemm(
    const float* __restrict__ accbuf, const float* __restrict__ W4,
    const float* __restrict__ b4,
    float* trans_io, float* gate_io, int ntiles)
{
    __shared__ __bf16 ldsW[DIM * LDSW_STRIDE];
    const int tid = threadIdx.x;
    stage_weights(W4, ldsW, tid);

    const int wave = tid >> 5;
    const int lane = tid & 31;
    const int tile = blockIdx.x * WAVES_PER_BLOCK + wave;
    if (tile >= ntiles) return;

    const int m0    = tile * 16;
    const int nlane = lane & 15;
    const int kg    = lane >> 4;

    bf16x16 a[4];  // A = relu(acc) in bf16, all K preloaded
    load_afrags(accbuf + (size_t)(m0 + nlane) * DIM + kg * 8, a, true);

    v8f acc[8] = {};
#pragma unroll
    for (int kb = 0; kb < 4; ++kb) {
#pragma unroll
        for (int nb = 0; nb < 8; ++nb) {
            bf16x16 b = load_bfrag(ldsW, nb, kb, nlane, kg);
            acc[nb] = __builtin_amdgcn_wmma_f32_16x16x32_bf16(
                false, a[kb], false, b, (short)0, acc[nb], false, false);
        }
    }

#pragma unroll
    for (int nb = 0; nb < 8; ++nb) {
        const int col = nb * 16 + nlane;
        const float bv = b4[col];
#pragma unroll
        for (int r = 0; r < 8; ++r) {
            const int m = r + (kg << 3);
            const size_t idx = (size_t)(m0 + m) * DIM + col;
            const float g2 = acc[nb][r] + bv;
            const float g1 = gate_io[idx];
            const float g  = sigmoidf_fast(g1 + g2);
            const float ov = fmaxf(accbuf[idx], 0.0f);   // output element
            const float tr = trans_io[idx];
            trans_io[idx] = ov + g * (tr - ov);          // out1
            gate_io[idx]  = tr + g * (ov - tr);          // out2
        }
    }
}

// ---------------------------------------------------------------------------
extern "C" void kernel_launch(void* const* d_in, const int* in_sizes, int n_in,
                              void* d_out, int out_size, void* d_ws, size_t ws_size,
                              hipStream_t stream) {
    const float* x    = (const float*)d_in[0];
    const float* res  = (const float*)d_in[1];
    const int*   arow = (const int*)  d_in[2];
    const int*   acol = (const int*)  d_in[3];
    const float* avls = (const float*)d_in[4];
    const float* w1   = (const float*)d_in[5];
    const float* w2   = (const float*)d_in[6];
    const float* w3   = (const float*)d_in[7];
    const float* w4   = (const float*)d_in[8];
    const float* b1   = (const float*)d_in[9];
    const float* b2   = (const float*)d_in[10];
    const float* b3   = (const float*)d_in[11];
    const float* b4   = (const float*)d_in[12];
    const float* eps  = (const float*)d_in[13];

    const int N = in_sizes[0] / DIM;        // 100000 (multiple of 16)
    const int E = in_sizes[2];              // 3200000
    const size_t NN = (size_t)N * DIM;

    float* support = (float*)d_ws;          // NN floats
    float* accbuf  = support + NN;          // NN floats
    float* out1    = (float*)d_out;         // trans scratch -> out1
    float* out2    = out1 + NN;             // gate1 scratch -> out2

    const int ntiles  = N / 16;
    const int gblocks = (ntiles + WAVES_PER_BLOCK - 1) / WAVES_PER_BLOCK;
    const dim3 blk(256);

    // support = x@w1 ; acc = eps*support + b1
    gemm_node<<<gblocks, blk, 0, stream>>>(x,   w1, b1, eps, support, accbuf, ntiles, 0);
    // trans = sigmoid(res@w2 + b2)  (into d_out lo half)
    gemm_node<<<gblocks, blk, 0, stream>>>(res, w2, b2, eps, out1, nullptr, ntiles, 1);
    // gate1 = x@w3 + b3             (into d_out hi half)
    gemm_node<<<gblocks, blk, 0, stream>>>(x,   w3, b3, eps, out2, nullptr, ntiles, 2);

    // acc += A @ support
    const long nwaves = ((long)E + EPW - 1) / EPW;
    const int  sblocks = (int)((nwaves + WAVES_PER_BLOCK - 1) / WAVES_PER_BLOCK);
    spmm_coo<<<sblocks, blk, 0, stream>>>(arow, acol, avls, support, accbuf, E);

    // output=relu(acc); gate2 GEMM; gated blend -> d_out
    finalize_gemm<<<gblocks, blk, 0, stream>>>(accbuf, w4, b4, out1, out2, ntiles);
}
